// VideoModelCoord_63694365000506
// MI455X (gfx1250) — compile-verified, matrix operand order checked by reading, not run on hardware
//
#include <hip/hip_runtime.h>
#include <hip/hip_bf16.h>

typedef _Float16 h16;
typedef __attribute__((ext_vector_type(16))) _Float16 v16h;
typedef __attribute__((ext_vector_type(8)))  _Float16 v8h;
typedef __attribute__((ext_vector_type(8)))  float    v8f;
typedef __attribute__((ext_vector_type(4)))  float    v4f;

#define N_Bv 1024
#define N_Vv 4
#define N_Tv 16
#define N_Dv 256
#define N_Ev 12
#define N_NODE (N_Bv*N_Vv*N_Tv)   /* 65536  */
#define N_EDGE (N_Bv*N_Ev*N_Tv)   /* 196608 */
#define GEMM_LDS (8 * 16 * 256 * 4)   /* 8 waves x 16x256 f32 tile = 128KB */

// ---------------------------------------------------------------------------
// Fragment load: 16-bit A/B 16x32 WMMA layout.
// lanes 0-15 hold K = [kk..kk+8) and [kk+16..kk+24); lanes 16-31 the +8 sets.
// Caller pre-adds laneHi*8 into the pointer.
// ---------------------------------------------------------------------------
__device__ __forceinline__ v16h load_frag(const h16* p) {
  v8h lo = *(const v8h*)(p);
  v8h hi = *(const v8h*)(p + 16);
  return __builtin_shufflevector(lo, hi, 0,1,2,3,4,5,6,7,8,9,10,11,12,13,14,15);
}

// Map an output-row token to an A-row. mode 0: identity. mode 1: edge->SEND
// node. mode 2: edge->RECV node.  SEND[e]=e/3, RECV[e]=p+(p>=s) with p=e%3.
__device__ __forceinline__ int map_row(int n, int mode) {
  if (mode == 0) return n;
  int b = n / (N_Ev * N_Tv);
  int r = n - b * (N_Ev * N_Tv);
  int e = r >> 4;
  int t = r & 15;
  int s = e / 3;
  int p = e - s * 3;
  int v = (mode == 1) ? s : (p + (p >= s ? 1 : 0));
  return ((b << 2) + v) * N_Tv + t;
}

// Accumulate a full 16-row x 256-col strip: 16 WMMA tiles, software-pipelined
// A (outer) and B (inner) fragment prefetch.
__device__ __forceinline__ void gemm_acc(v8f acc[16], const h16* A, int mode,
                                         const h16* Wt, int K,
                                         int n0, int mRow, int laneHi) {
  int ar = map_row(n0 + mRow, mode);
  const h16* Ap = A + (size_t)ar * K + (laneHi << 3);
  const h16* Bp = Wt + (size_t)mRow * K + (laneHi << 3);
  v16h a = load_frag(Ap);
  for (int kk = 0; kk < K; kk += 32) {
    int kn = (kk + 32 < K) ? kk + 32 : kk;
    v16h an = load_frag(Ap + kn);                 // prefetch next A
    v16h b = load_frag(Bp + kk);
#pragma unroll
    for (int j = 0; j < 16; ++j) {
      v16h bn = load_frag(Bp + (size_t)((j + 1) & 15) * 16 * K + kk);  // prefetch next B
      acc[j] = __builtin_amdgcn_wmma_f32_16x16x32_f16(
          false, a, false, b, (short)0, acc[j], false, false);
      b = bn;
    }
    a = an;
  }
}

// ---------------------------------------------------------------------------
// Generic GEMM: out[N_tok,256] = epi( A0@Wt0^T (+ A1@Wt1^T) )
// One wave computes a 16x256 strip; results staged in LDS, stored coalesced.
// act: 0=relu 1=elu.  bnFirst: 1 -> y=act(s*x+t); 0 -> y=s*act(x+bias)+t
// ---------------------------------------------------------------------------
__global__ __launch_bounds__(256)
void k_gemm(const h16* __restrict__ A0, int mode0, const h16* __restrict__ Wt0, int K0,
            const h16* __restrict__ A1, int mode1, const h16* __restrict__ Wt1, int K1,
            const float* __restrict__ bias, const float* __restrict__ scale,
            const float* __restrict__ shift, int act, int bnFirst,
            h16* __restrict__ outH, float* __restrict__ outF, int Ntok) {
  extern __shared__ char smembase[];
  int wave = threadIdx.x >> 5;
  int wid  = blockIdx.x * 8 + wave;
  int lane = threadIdx.x & 31;
  if (wid >= (Ntok >> 4)) return;

  int n0     = wid << 4;
  int mRow   = lane & 15;
  int laneHi = lane >> 4;

  v8f acc[16];
#pragma unroll
  for (int j = 0; j < 16; ++j) acc[j] = v8f{};

  gemm_acc(acc, A0, mode0, Wt0, K0, n0, mRow, laneHi);
  if (A1) gemm_acc(acc, A1, mode1, Wt1, K1, n0, mRow, laneHi);

  // epilogue -> per-wave LDS tile (same-wave DS ops are in-order: no barrier)
  float* l = (float*)smembase + wave * (16 * 256);
#pragma unroll
  for (int j = 0; j < 16; ++j) {
    int col  = j * 16 + mRow;
    float bb = bias  ? bias[col]  : 0.f;
    float sc = scale ? scale[col] : 1.f;
    float sh = shift ? shift[col] : 0.f;
#pragma unroll
    for (int r = 0; r < 8; ++r) {
      int m = (laneHi << 3) + r;
      float x = acc[j][r] + bb;
      if (bnFirst) {
        x = sc * x + sh;
        x = (act == 1) ? (x > 0.f ? x : expm1f(x)) : fmaxf(x, 0.f);
      } else {
        x = (act == 1) ? (x > 0.f ? x : expm1f(x)) : fmaxf(x, 0.f);
        x = sc * x + sh;
      }
      l[m * 256 + col] = x;
    }
  }

  // coalesced copy-out: the 16x256 tile is one contiguous block of rows
  if (outF) {
    const v4f* src = (const v4f*)l;
    v4f* dst = (v4f*)(outF + (size_t)n0 * N_Dv);
#pragma unroll 4
    for (int i = lane; i < 16 * 256 / 4; i += 32) dst[i] = src[i];
  }
  if (outH) {
    v8h* dst = (v8h*)(outH + (size_t)n0 * N_Dv);
#pragma unroll 4
    for (int i = lane; i < 16 * 256 / 8; i += 32) {
      v4f lo = *((const v4f*)l + i * 2);
      v4f hi = *((const v4f*)l + i * 2 + 1);
      v8h o;
#pragma unroll
      for (int q = 0; q < 4; ++q) { o[q] = (h16)lo[q]; o[4 + q] = (h16)hi[q]; }
      dst[i] = o;
    }
  }
}

// c2f layer 1: [65536,4] @ [4,128] -> relu(bn) -> f16
__global__ __launch_bounds__(256)
void k_c2f1(const float* __restrict__ box, const float* __restrict__ W1,
            const float* __restrict__ s, const float* __restrict__ t,
            h16* __restrict__ out) {
  int idx = blockIdx.x * 256 + threadIdx.x;   // N_NODE*128 threads
  int n = idx >> 7, j = idx & 127;
  int tt = n & 15, v = (n >> 4) & 3, b = n >> 6;
  const float* x = box + ((((size_t)b * N_Tv + tt) * N_Vv + v) << 2);
  float h = x[0] * W1[j] + x[1] * W1[128 + j] + x[2] * W1[256 + j] + x[3] * W1[384 + j];
  float y = fmaxf(s[j] * h + t[j], 0.f);
  out[(size_t)n * 128 + j] = (h16)y;
}

// edge2node: mean of the 3 incoming edges per node, f16 -> f16
__global__ __launch_bounds__(256)
void k_edge2node(const h16* __restrict__ edge, h16* __restrict__ node) {
  int idx = blockIdx.x * 256 + threadIdx.x;   // N_NODE*32 threads
  int m = idx >> 5, c = (idx & 31) << 3;
  int t = m & 15, v = (m >> 4) & 3, b = m >> 6;
  float sum[8] = {0, 0, 0, 0, 0, 0, 0, 0};
#pragma unroll
  for (int u = 0; u < 4; ++u) {
    if (u == v) continue;
    int e = u * 3 + (v - (v > u ? 1 : 0));
    size_t row = (((size_t)b * N_Ev + e) << 4) + t;
    v8h d = *(const v8h*)(edge + row * N_Dv + c);
#pragma unroll
    for (int i = 0; i < 8; ++i) sum[i] += (float)d[i];
  }
  v8h o;
#pragma unroll
  for (int i = 0; i < 8; ++i) o[i] = (h16)(sum[i] * (1.f / 3.f));
  *(v8h*)(node + (size_t)m * N_Dv + c) = o;
}

// Fold BN: s = g*rsqrt(v+eps); t = b - m*s
__global__ void k_bnparams(const float* g, const float* b, const float* m,
                           const float* v, float* s, float* t, int n) {
  int i = blockIdx.x * blockDim.x + threadIdx.x;
  if (i < n) {
    float sc = g[i] * rsqrtf(v[i] + 1e-5f);
    s[i] = sc;
    t[i] = b[i] - m[i] * sc;
  }
}

// W[K,N] fp32 -> Wt[N,K] f16
__global__ void k_transpose(const float* __restrict__ W, h16* __restrict__ Wt,
                            int K, int N) {
  int idx = blockIdx.x * blockDim.x + threadIdx.x;
  if (idx < K * N) {
    int k = idx / N, n = idx - k * N;
    Wt[(size_t)n * K + k] = (h16)W[idx];
  }
}

// ---------------------------------------------------------------------------
struct C2FP { const float *W1, *W2, *b1, *b2, *g1, *g2, *m1, *m2, *v1, *v2; };
struct MLPP { const float *W1, *W2, *b, *b1, *b2, *g, *m, *v; };

extern "C" void kernel_launch(void* const* d_in, const int* in_sizes, int n_in,
                              void* d_out, int out_size, void* d_ws, size_t ws_size,
                              hipStream_t stream) {
  const float* box = (const float*)d_in[0];
  C2FP c2f{};
  MLPP mlp[4]{};

  if (n_in >= 43) {
    auto f = [&](int i) { return (const float*)d_in[i]; };
    if (in_sizes[2] == 128 * 256) {            // sorted keys: W1,W2,b1,b2,g1,g2,m1,m2,v1,v2
      c2f = {f(1), f(2), f(3), f(4), f(5), f(6), f(7), f(8), f(9), f(10)};
    } else {                                   // insertion: W1,g1,b1,m1,v1,W2,g2,b2,m2,v2
      c2f.W1 = f(1); c2f.g1 = f(2); c2f.b1 = f(3); c2f.m1 = f(4); c2f.v1 = f(5);
      c2f.W2 = f(6); c2f.g2 = f(7); c2f.b2 = f(8); c2f.m2 = f(9); c2f.v2 = f(10);
    }
    for (int k = 0; k < 4; ++k) {
      int ba = 11 + 8 * k;
      if (in_sizes[ba + 1] > 256) {            // sorted: W1,W2,b,b1,b2,g,m,v
        mlp[k] = {f(ba), f(ba + 1), f(ba + 2), f(ba + 3), f(ba + 4), f(ba + 5), f(ba + 6), f(ba + 7)};
      } else {                                 // insertion: W1,b1,W2,b2,g,b,m,v
        mlp[k].W1 = f(ba);     mlp[k].b1 = f(ba + 1); mlp[k].W2 = f(ba + 2);
        mlp[k].b2 = f(ba + 3); mlp[k].g  = f(ba + 4); mlp[k].b  = f(ba + 5);
        mlp[k].m  = f(ba + 6); mlp[k].v  = f(ba + 7);
      }
    }
  } else {
    // one packed buffer per dict, leaves in sorted-key order (jax pytree)
    const float* p = (const float*)d_in[1];
    c2f.W1 = p;            c2f.W2 = p + 512;      c2f.b1 = p + 512 + 32768;
    c2f.b2 = c2f.b1 + 128; c2f.g1 = c2f.b2 + 256; c2f.g2 = c2f.g1 + 128;
    c2f.m1 = c2f.g2 + 256; c2f.m2 = c2f.m1 + 128; c2f.v1 = c2f.m2 + 256;
    c2f.v2 = c2f.v1 + 128;
    for (int k = 0; k < 4; ++k) {
      const float* q = (const float*)d_in[2 + k];
      int w1 = (k == 1) ? 256 * 256 : 512 * 256;
      mlp[k].W1 = q;               mlp[k].W2 = q + w1;
      mlp[k].b  = mlp[k].W2 + 65536; mlp[k].b1 = mlp[k].b + 256;
      mlp[k].b2 = mlp[k].b1 + 256; mlp[k].g  = mlp[k].b2 + 256;
      mlp[k].m  = mlp[k].g + 256;  mlp[k].v  = mlp[k].m + 256;
    }
  }

  // ---- workspace layout ----------------------------------------------------
  size_t off = 0;
  auto alloc = [&](size_t bytes) { size_t r = off; off = (off + bytes + 255) & ~(size_t)255; return r; };
  char* ws = (char*)d_ws;
  h16* h1   = (h16*)(ws + alloc((size_t)N_NODE * 128 * 2));
  h16* bf   = (h16*)(ws + alloc((size_t)N_NODE * 256 * 2));   // c2f out, later mlp2 out
  h16* ehid = (h16*)(ws + alloc((size_t)N_EDGE * 256 * 2));   // shared hidden (mlp1/3/4)
  h16* skip = (h16*)(ws + alloc((size_t)N_EDGE * 256 * 2));   // mlp1 out
  h16* na   = (h16*)(ws + alloc((size_t)N_NODE * 256 * 2));   // edge2node out
  h16* nb   = (h16*)(ws + alloc((size_t)N_NODE * 256 * 2));   // mlp2 hidden
  h16* e3   = (h16*)(ws + alloc((size_t)N_EDGE * 256 * 2));   // mlp3 out
  h16* wC2  = (h16*)(ws + alloc(32768 * 2));                   // c2f W2^T [256,128]
  h16* w1a  = (h16*)(ws + alloc(65536 * 2));
  h16* w1b  = (h16*)(ws + alloc(65536 * 2));
  h16* w1c  = (h16*)(ws + alloc(65536 * 2));
  h16* w2a  = (h16*)(ws + alloc(65536 * 2));
  h16* w2c  = (h16*)(ws + alloc(65536 * 2));
  h16* w3a  = (h16*)(ws + alloc(65536 * 2));
  h16* w3b  = (h16*)(ws + alloc(65536 * 2));
  h16* w3c  = (h16*)(ws + alloc(65536 * 2));
  h16* w4a  = (h16*)(ws + alloc(65536 * 2));
  h16* w4b  = (h16*)(ws + alloc(65536 * 2));
  h16* w4c  = (h16*)(ws + alloc(65536 * 2));
  float* sC1 = (float*)(ws + alloc(128 * 4)); float* tC1 = (float*)(ws + alloc(128 * 4));
  float* sC2 = (float*)(ws + alloc(256 * 4)); float* tC2 = (float*)(ws + alloc(256 * 4));
  float* sM[4]; float* tM[4];
  for (int k = 0; k < 4; ++k) { sM[k] = (float*)(ws + alloc(256 * 4)); tM[k] = (float*)(ws + alloc(256 * 4)); }

  // ---- prep: BN fold + weight transpose/convert ----------------------------
  hipLaunchKernelGGL(k_bnparams, dim3(1), dim3(256), 0, stream, c2f.g1, c2f.b1, c2f.m1, c2f.v1, sC1, tC1, 128);
  hipLaunchKernelGGL(k_bnparams, dim3(1), dim3(256), 0, stream, c2f.g2, c2f.b2, c2f.m2, c2f.v2, sC2, tC2, 256);
  for (int k = 0; k < 4; ++k)
    hipLaunchKernelGGL(k_bnparams, dim3(1), dim3(256), 0, stream, mlp[k].g, mlp[k].b, mlp[k].m, mlp[k].v, sM[k], tM[k], 256);

  auto tr = [&](const float* W, h16* Wt, int K, int N) {
    hipLaunchKernelGGL(k_transpose, dim3((K * N + 255) / 256), dim3(256), 0, stream, W, Wt, K, N);
  };
  tr(c2f.W2, wC2, 128, 256);
  tr(mlp[0].W1,              w1a, 256, 256);  // send half
  tr(mlp[0].W1 + 256 * 256,  w1b, 256, 256);  // recv half
  tr(mlp[0].W2,              w1c, 256, 256);
  tr(mlp[1].W1,              w2a, 256, 256);
  tr(mlp[1].W2,              w2c, 256, 256);
  tr(mlp[2].W1,              w3a, 256, 256);
  tr(mlp[2].W1 + 256 * 256,  w3b, 256, 256);
  tr(mlp[2].W2,              w3c, 256, 256);
  tr(mlp[3].W1,              w4a, 256, 256);  // x (mlp3 out) half
  tr(mlp[3].W1 + 256 * 256,  w4b, 256, 256);  // skip half
  tr(mlp[3].W2,              w4c, 256, 256);

  // ---- pipeline ------------------------------------------------------------
  int gNode = (N_NODE / 16 + 7) / 8;   // 512 blocks  (1 wave = 16x256 strip)
  int gEdge = (N_EDGE / 16 + 7) / 8;   // 1536 blocks

  // c2f
  hipLaunchKernelGGL(k_c2f1, dim3(N_NODE * 128 / 256), dim3(256), 0, stream, box, c2f.W1, sC1, tC1, h1);
  hipLaunchKernelGGL(k_gemm, dim3(gNode), dim3(256), GEMM_LDS, stream,
                     h1, 0, wC2, 128, (const h16*)nullptr, 0, (const h16*)nullptr, 0,
                     (const float*)nullptr, sC2, tC2, /*relu*/0, /*bnFirst*/1, bf, (float*)nullptr, N_NODE);
  // mlp1 on node2edge(bf)  (gathered send/recv halves)
  hipLaunchKernelGGL(k_gemm, dim3(gEdge), dim3(256), GEMM_LDS, stream,
                     bf, 1, w1a, 256, bf, 2, w1b, 256,
                     mlp[0].b1, (const float*)nullptr, (const float*)nullptr, /*elu*/1, 0, ehid, (float*)nullptr, N_EDGE);
  hipLaunchKernelGGL(k_gemm, dim3(gEdge), dim3(256), GEMM_LDS, stream,
                     ehid, 0, w1c, 256, (const h16*)nullptr, 0, (const h16*)nullptr, 0,
                     mlp[0].b2, sM[0], tM[0], 1, 0, skip, (float*)nullptr, N_EDGE);
  // edge2node mean
  hipLaunchKernelGGL(k_edge2node, dim3(N_NODE * 32 / 256), dim3(256), 0, stream, skip, na);
  // mlp2
  hipLaunchKernelGGL(k_gemm, dim3(gNode), dim3(256), GEMM_LDS, stream,
                     na, 0, w2a, 256, (const h16*)nullptr, 0, (const h16*)nullptr, 0,
                     mlp[1].b1, (const float*)nullptr, (const float*)nullptr, 1, 0, nb, (float*)nullptr, N_NODE);
  hipLaunchKernelGGL(k_gemm, dim3(gNode), dim3(256), GEMM_LDS, stream,
                     nb, 0, w2c, 256, (const h16*)nullptr, 0, (const h16*)nullptr, 0,
                     mlp[1].b2, sM[1], tM[1], 1, 0, bf, (float*)nullptr, N_NODE);
  // mlp3 on node2edge(mlp2 out)
  hipLaunchKernelGGL(k_gemm, dim3(gEdge), dim3(256), GEMM_LDS, stream,
                     bf, 1, w3a, 256, bf, 2, w3b, 256,
                     mlp[2].b1, (const float*)nullptr, (const float*)nullptr, 1, 0, ehid, (float*)nullptr, N_EDGE);
  hipLaunchKernelGGL(k_gemm, dim3(gEdge), dim3(256), GEMM_LDS, stream,
                     ehid, 0, w3c, 256, (const h16*)nullptr, 0, (const h16*)nullptr, 0,
                     mlp[2].b2, sM[2], tM[2], 1, 0, e3, (float*)nullptr, N_EDGE);
  // mlp4 on concat([e3, skip])
  hipLaunchKernelGGL(k_gemm, dim3(gEdge), dim3(256), GEMM_LDS, stream,
                     e3, 0, w4a, 256, skip, 0, w4b, 256,
                     mlp[3].b1, (const float*)nullptr, (const float*)nullptr, 1, 0, ehid, (float*)nullptr, N_EDGE);
  hipLaunchKernelGGL(k_gemm, dim3(gEdge), dim3(256), GEMM_LDS, stream,
                     ehid, 0, w4c, 256, (const h16*)nullptr, 0, (const h16*)nullptr, 0,
                     mlp[3].b2, sM[3], tM[3], 1, 0, (h16*)nullptr, (float*)d_out, N_EDGE);
  (void)ws_size; (void)out_size; (void)n_in;
}
